// TorchLocalAttention_5669356834645
// MI455X (gfx1250) — matrix operand order repaired and checked by reading, not compiled, not run on hardware
//
#include <hip/hip_runtime.h>
#include <hip/hip_bf16.h>

// CDNA5 (gfx1250) wave32 WMMA types
typedef __attribute__((ext_vector_type(16))) __bf16 v16bf;
typedef __attribute__((ext_vector_type(8)))  float  v8f;

union BfFrag { v16bf v; unsigned short u[16]; uint4 q[2]; };

__device__ __forceinline__ unsigned short f2bf(float f) {
  unsigned int x = __float_as_uint(f);
  unsigned int r = x + 0x7FFFu + ((x >> 16) & 1u);   // round-to-nearest-even
  return (unsigned short)(r >> 16);
}

#define CCH  128
#define HH   64
#define WW   64
#define NIMG 8
#define HW   4096   // 64*64
#define HP   70     // padded rows (halo 3)
#define WP   80     // padded cols (halo 3 left, 13 right: keeps b128 band reads in-bounds)

// ---------------------------------------------------------------------------
// zero-fill the padded key/value workspaces (borders must be 0 every call)
// ---------------------------------------------------------------------------
__global__ void zero_ws_kernel(unsigned int* __restrict__ p, int n) {
  int i = blockIdx.x * blockDim.x + threadIdx.x;
  int stride = gridDim.x * blockDim.x;
  for (; i < n; i += stride) p[i] = 0u;
}

// convert the three 128x128 f32 weight matrices to bf16 once per call
__global__ void prep_wb_kernel(const float* __restrict__ w1,
                               const float* __restrict__ w2,
                               const float* __restrict__ w3,
                               unsigned short* __restrict__ wb) {
  int i = blockIdx.x * blockDim.x + threadIdx.x;   // 0..49151
  if (i < 16384)      wb[i] = f2bf(w1[i]);
  else if (i < 32768) wb[i] = f2bf(w2[i - 16384]);
  else if (i < 49152) wb[i] = f2bf(w3[i - 32768]);
}

// ---------------------------------------------------------------------------
// Stage 1: three 1x1 convs as bf16 WMMA GEMMs.  Y_m[o,q] = sum_c Wm[o,c] X[c,q]
// Block: one image n, one 64-pixel row. 8 waves; wave = 16-row o-tile.
// All fragments are two 16-byte runs (k 8h.., 16+8h..) -> b128 loads, no repacking.
// ---------------------------------------------------------------------------
__global__ __launch_bounds__(256)
void conv3_gemm_kernel(const float* __restrict__ x,
                       const unsigned short* __restrict__ wb,  // [3][128][128] bf16
                       unsigned short* __restrict__ x1b,
                       unsigned short* __restrict__ x2b,
                       unsigned short* __restrict__ x3b) {
  __shared__ __align__(16) unsigned short Xb[64 * CCH];   // [q_local][c] bf16, 16 KB
  const int n  = blockIdx.y;
  const int q0 = blockIdx.x * 64;
  const int tid = threadIdx.x;

  const int wid  = tid >> 5;
  const int lane = tid & 31;
  const int hsel = lane >> 4;   // lane-half select per ISA VGPR layout
  const int l16  = lane & 15;
  const int o0   = wid * 16;
  const int mrow = o0 + l16;

  // warm L2/WGP$ for this wave's weight rows while we stage the X tile
#pragma unroll
  for (int m = 0; m < 3; ++m)
    __builtin_prefetch(wb + m * 16384 + mrow * CCH, 0, 3);

  // stage activation tile to LDS transposed [q][c] (coalesced float4 reads)
  for (int idx = tid; idx < CCH * 16; idx += 256) {
    int c = idx >> 4, j4 = (idx & 15) << 2;
    const float4 f = *(const float4*)(x + (size_t)(n * CCH + c) * HW + q0 + j4);
    Xb[(j4 + 0) * CCH + c] = f2bf(f.x);
    Xb[(j4 + 1) * CCH + c] = f2bf(f.y);
    Xb[(j4 + 2) * CCH + c] = f2bf(f.z);
    Xb[(j4 + 3) * CCH + c] = f2bf(f.w);
  }
  __syncthreads();

  // A fragments: 3 weights x 4 K-slabs, each = two b128 loads of a bf16 row
  BfFrag aw[3][4];
#pragma unroll
  for (int m = 0; m < 3; ++m)
#pragma unroll
    for (int ks = 0; ks < 4; ++ks) {
      const unsigned short* wr = wb + m * 16384 + mrow * CCH + ks * 32;
      aw[m][ks].q[0] = *(const uint4*)(wr + 8 * hsel);        // K = ks*32+8h .. +7
      aw[m][ks].q[1] = *(const uint4*)(wr + 16 + 8 * hsel);   // K = ks*32+16+8h .. +7
    }

  for (int qt = 0; qt < 4; ++qt) {
    const int col = qt * 16 + l16;
    BfFrag bx[4];
#pragma unroll
    for (int ks = 0; ks < 4; ++ks) {
      const unsigned short* xr = &Xb[col * CCH + ks * 32 + 8 * hsel];
      bx[ks].q[0] = *(const uint4*)(xr);        // K = ks*32+8h .. +7
      bx[ks].q[1] = *(const uint4*)(xr + 16);   // K = ks*32+16+8h .. +7
    }
#pragma unroll
    for (int m = 0; m < 3; ++m) {
      v8f acc = {};
#pragma unroll
      for (int ks = 0; ks < 4; ++ks)
        acc = __builtin_amdgcn_wmma_f32_16x16x32_bf16(
            false, aw[m][ks].v, false, bx[ks].v, (short)0, acc, false, false);
      // C/D layout: lane l, reg r -> (M = r + 8*hsel, N = l16)
#pragma unroll
      for (int r = 0; r < 8; ++r) {
        int o = o0 + r + 8 * hsel;
        int q = q0 + qt * 16 + l16;
        unsigned short bv = f2bf(acc[r]);
        if (m == 0) {
          x1b[(size_t)(n * CCH + o) * HW + q] = bv;
        } else {
          int hh = q >> 6, wc = q & 63;
          unsigned short* dst = (m == 1) ? x2b : x3b;
          dst[((size_t)(n * CCH + o) * HP + hh + 3) * WP + wc + 3] = bv;
        }
      }
    }
  }
}

// ---------------------------------------------------------------------------
// Stage 2: 7x7 local attention for a 16-query row tile.
// ---------------------------------------------------------------------------
__global__ __launch_bounds__(256)
void local_attn_kernel(const unsigned short* __restrict__ x1b,
                       const unsigned short* __restrict__ x2b,
                       const unsigned short* __restrict__ x3b,
                       float* __restrict__ out) {
  __shared__ __align__(16) unsigned short ldsQ[16 * CCH];       // [m][c]     4 KB
  __shared__ __align__(16) unsigned short ldsK[7 * 22 * CCH];   // [di][j][c] 38.5 KB
  __shared__ __align__(16) float          simS[7 * 16 * 22];    // [di][m][j]  9.6 KB
  __shared__ __align__(16) unsigned short attnB[7 * 16 * 32];   // [di][m][k]  7 KB

  const int w0 = blockIdx.x * 16;
  const int h  = blockIdx.y;
  const int n  = blockIdx.z;
  const int tid = threadIdx.x;

  const int wid  = tid >> 5;
  const int lane = tid & 31;
  const int hsel = lane >> 4;
  const int l16  = lane & 15;
  const int c0   = wid * 16;

  // prefetch this wave's V-band rows: consumed only after sim+softmax, so the
  // L2 -> WGP fill overlaps the WMMA phase (global_prefetch_b8)
#pragma unroll
  for (int di = 0; di < 7; ++di)
    __builtin_prefetch(
        x3b + ((size_t)(n * CCH + c0 + l16) * HP + h + di) * WP + w0, 0, 3);

  // Q tile -> LDS transposed [m][c] (b128 global reads, unpack once)
  for (int idx = tid; idx < CCH * 2; idx += 256) {
    int c = idx >> 1, g = (idx & 1) * 8;
    uint4 d = *(const uint4*)(x1b + (size_t)(n * CCH + c) * HW + h * WW + w0 + g);
    unsigned int dv[4] = {d.x, d.y, d.z, d.w};
#pragma unroll
    for (int t = 0; t < 4; ++t) {
      ldsQ[(g + 2 * t + 0) * CCH + c] = (unsigned short)dv[t];
      ldsQ[(g + 2 * t + 1) * CCH + c] = (unsigned short)(dv[t] >> 16);
    }
  }
  // key band -> LDS [di][j][c]; di-major loop so all index math is shifts,
  // inner LDS offsets are compile-time immediates
  for (int p = tid; p < 7 * CCH; p += 256) {
    int di = p >> 7, c = p & 127;
    const unsigned short* src =
        x2b + ((size_t)(n * CCH + c) * HP + h + di) * WP + w0;
    unsigned short* dst = &ldsK[(di * 22) * CCH + c];
#pragma unroll
    for (int j2 = 0; j2 < 11; ++j2) {
      unsigned int d = *(const unsigned int*)(src + 2 * j2);
      dst[(2 * j2 + 0) * CCH] = (unsigned short)d;
      dst[(2 * j2 + 1) * CCH] = (unsigned short)(d >> 16);
    }
  }
  for (int idx = tid; idx < 7 * 16 * 32 / 2; idx += 256)
    ((unsigned int*)attnB)[idx] = 0u;
  __syncthreads();

  // --- similarity: S[di][m][j] = sum_c Q[c,m] * K[c,di,j], j in [0,22) ---
  if (wid < 7) {
    const int di = wid;
#pragma unroll
    for (int tile = 0; tile < 2; ++tile) {
      const int j  = tile * 16 + l16;
      const int jc = (j < 22) ? j : 21;   // lanes j>=22 compute discarded columns
      v8f acc = {};
#pragma unroll
      for (int ks = 0; ks < 4; ++ks) {
        BfFrag a, b;
        const unsigned short* qr = &ldsQ[l16 * CCH + ks * 32];
        a.q[0] = *(const uint4*)(qr + 8 * hsel);
        a.q[1] = *(const uint4*)(qr + 16 + 8 * hsel);
        const unsigned short* kr = &ldsK[(di * 22 + jc) * CCH + ks * 32 + 8 * hsel];
        b.q[0] = *(const uint4*)(kr);
        b.q[1] = *(const uint4*)(kr + 16);
        acc = __builtin_amdgcn_wmma_f32_16x16x32_bf16(
            false, a.v, false, b.v, (short)0, acc, false, false);
      }
      if (j < 22) {
#pragma unroll
        for (int r = 0; r < 8; ++r) {
          int m = r + 8 * hsel;
          simS[(di * 16 + m) * 22 + j] = acc[r];
        }
      }
    }
  }
  __syncthreads();

  // --- softmax over 49 window scores; scatter into banded bf16 B matrices ---
  if (tid < 16) {
    const int m = tid;
    float v[49], mx = -3.0e38f;
#pragma unroll
    for (int di = 0; di < 7; ++di)
#pragma unroll
      for (int dj = 0; dj < 7; ++dj) {
        float s = simS[(di * 16 + m) * 22 + m + dj];
        v[di * 7 + dj] = s;
        mx = fmaxf(mx, s);
      }
    float sum = 0.f;
#pragma unroll
    for (int k = 0; k < 49; ++k) { v[k] = __expf(v[k] - mx); sum += v[k]; }
    float inv = 1.0f / sum;
#pragma unroll
    for (int di = 0; di < 7; ++di)
#pragma unroll
      for (int dj = 0; dj < 7; ++dj)
        attnB[(di * 16 + m) * 32 + m + dj] = f2bf(v[di * 7 + dj] * inv);
  }
  __syncthreads();

  // --- aggregation: out[c0+mc, m] = sum_di V_band (16c x 32) x attn (32 x 16q)
  // A rows beyond band pos 21 hold real-but-unused data; the matching attnB
  // K-rows are zero, so those products vanish. WP=80 keeps b128 reads in-bounds.
  {
    v8f acc = {};
#pragma unroll
    for (int di = 0; di < 7; ++di) {
      BfFrag a, b;
      const unsigned short* vr =
          x3b + ((size_t)(n * CCH + c0 + l16) * HP + h + di) * WP + w0;
      a.q[0] = *(const uint4*)(vr + 8 * hsel);        // band 8h .. 8h+7
      a.q[1] = *(const uint4*)(vr + 16 + 8 * hsel);   // band 16+8h .. +7
      const unsigned short* br = &attnB[(di * 16 + l16) * 32 + 8 * hsel];
      b.q[0] = *(const uint4*)(br);
      b.q[1] = *(const uint4*)(br + 16);
      acc = __builtin_amdgcn_wmma_f32_16x16x32_bf16(
          false, a.v, false, b.v, (short)0, acc, false, false);
    }
#pragma unroll
    for (int r = 0; r < 8; ++r) {
      int c = c0 + r + 8 * hsel;
      out[(size_t)(n * CCH + c) * HW + h * WW + w0 + l16] = acc[r];
    }
  }
}

// ---------------------------------------------------------------------------
extern "C" void kernel_launch(void* const* d_in, const int* in_sizes, int n_in,
                              void* d_out, int out_size, void* d_ws, size_t ws_size,
                              hipStream_t stream) {
  const float* x  = (const float*)d_in[0];
  const float* w1 = (const float*)d_in[1];
  const float* w2 = (const float*)d_in[2];
  const float* w3 = (const float*)d_in[3];
  float* out = (float*)d_out;

  char* ws = (char*)d_ws;
  const size_t x1_bytes  = (size_t)NIMG * CCH * HW * 2;        //  8,388,608
  const size_t pad_bytes = (size_t)NIMG * CCH * HP * WP * 2;   // 11,468,800
  unsigned short* x1b = (unsigned short*)ws;
  unsigned short* x2b = (unsigned short*)(ws + x1_bytes);
  unsigned short* x3b = (unsigned short*)(ws + x1_bytes + pad_bytes);
  unsigned short* wb  = (unsigned short*)(ws + x1_bytes + 2 * pad_bytes);

  // zero both padded buffers (x2b,x3b contiguous) so halos are 0 every call
  int nzero = (int)((2 * pad_bytes) / 4);
  zero_ws_kernel<<<2048, 256, 0, stream>>>((unsigned int*)x2b, nzero);
  prep_wb_kernel<<<192, 256, 0, stream>>>(w1, w2, w3, wb);

  conv3_gemm_kernel<<<dim3(64, NIMG), 256, 0, stream>>>(x, wb, x1b, x2b, x3b);

  local_attn_kernel<<<dim3(4, HH, NIMG), 256, 0, stream>>>(x1b, x2b, x3b, out);
}
